// LSTMCell_84490596647260
// MI455X (gfx1250) — matrix operand order repaired and checked by reading, not compiled, not run on hardware
//
#include <hip/hip_runtime.h>

#define B_DIM  8192
#define IN_DIM 1024
#define H_DIM  1024
#define LDT    40   // LDS row stride in bf16 elements (80 bytes, 16B-aligned)

typedef __attribute__((ext_vector_type(16))) __bf16 v16bf;
typedef __attribute__((ext_vector_type(8)))  __bf16 v8bf;
typedef __attribute__((ext_vector_type(4)))  __bf16 v4bf;
typedef __attribute__((ext_vector_type(8)))  float  v8f;

// A fragment (16x32, M x K): lane m holds logical K {0..7,16..23} (lanes<16)
// or {8..15,24..31} (lanes>=16)  -> two b128 LDS loads.
__device__ __forceinline__ v16bf frag_a(const __bf16* base, int mBase, int lane) {
    const int m = mBase + (lane & 15);
    const __bf16* p = base + m * LDT + ((lane < 16) ? 0 : 8);
    union { v16bf f; v8bf h[2]; } u;
    u.h[0] = *(const v8bf*)(p);
    u.h[1] = *(const v8bf*)(p + 16);
    return u.f;
}
// B fragment (32x16, K x N): lane n holds logical K {0..7,16..23} (lanes<16) or
// {8..15,24..31} (lanes>=16), from a row of W -> two b128 LDS loads.
__device__ __forceinline__ v16bf frag_b(const __bf16* base, int nBase, int lane) {
    const int n = nBase + (lane & 15);
    const __bf16* p = base + n * LDT + ((lane < 16) ? 0 : 8);
    union { v16bf f; v8bf h[2]; } u;
    u.h[0] = *(const v8bf*)(p);
    u.h[1] = *(const v8bf*)(p + 16);
    return u.f;
}

__device__ __forceinline__ void cvt_store4(__bf16* dst, float4 v) {
    v4bf b;
    b[0] = (__bf16)v.x; b[1] = (__bf16)v.y; b[2] = (__bf16)v.z; b[3] = (__bf16)v.w;
    *(v4bf*)dst = b;
}

__device__ __forceinline__ float sigmoid_fast(float x) {
    return 1.0f / (1.0f + __expf(-x));
}
__device__ __forceinline__ float tanh_fast(float x) {
    const float e = __expf(-2.0f * x);
    return (1.0f - e) / (1.0f + e);
}

__global__ __launch_bounds__(256)
void lstm_fused_wmma(
    const float* __restrict__ x, const float* __restrict__ hPrev, const float* __restrict__ cPrev,
    const float* __restrict__ Wx0, const float* __restrict__ Wx1,
    const float* __restrict__ Wx2, const float* __restrict__ Wx3,
    const float* __restrict__ Wh0, const float* __restrict__ Wh1,
    const float* __restrict__ Wh2, const float* __restrict__ Wh3,
    const float* __restrict__ bi, const float* __restrict__ bf_,
    const float* __restrict__ bg, const float* __restrict__ bo,
    float* __restrict__ out)
{
    __shared__ __align__(16) __bf16 sA[2][128][LDT];     // 20 KB : A tile (bf16)
    __shared__ __align__(16) __bf16 sW[2][4][64][LDT];   // 40 KB : 4 gate W tiles

    const int tid  = threadIdx.x;
    const int lane = tid & 31;
    const int wid  = tid >> 5;
    const int wm   = wid >> 1;                 // 0..3 : wave row
    const int wn   = wid & 1;                  // 0..1 : wave col

    const long rowM0 = (long)blockIdx.x * 128;
    const int  colN0 = blockIdx.y * 64;

    const v8f vzero = {};
    v8f acc[4][4];                             // [gate][2x2 16x16 subtiles]
    #pragma unroll
    for (int g = 0; g < 4; ++g)
        #pragma unroll
        for (int q = 0; q < 4; ++q)
            acc[g][q] = vzero;

    auto load_tile = [&](int buf, int step) {
        const int p  = step >> 5;              // phase 0: x * Wx, phase 1: h * Wh
        const int k0 = (step & 31) * 32;
        const float* Asrc = p ? hPrev : x;
        // A tile: 128 x 32 fp32 -> bf16 (1024 float4 chunks / 256 threads)
        #pragma unroll
        for (int i = 0; i < 4; ++i) {
            const int idx = tid + i * 256;
            const int r   = idx >> 3;
            const int c4  = idx & 7;
            float4 v = *(const float4*)(Asrc + (rowM0 + r) * (long)IN_DIM + k0 + c4 * 4);
            cvt_store4(&sA[buf][r][c4 * 4], v);
        }
        // W tiles: 4 gates x 64 x 32
        #pragma unroll
        for (int g = 0; g < 4; ++g) {
            const float* Wsrc =
                (g == 0) ? (p ? Wh0 : Wx0) :
                (g == 1) ? (p ? Wh1 : Wx1) :
                (g == 2) ? (p ? Wh2 : Wx2) :
                           (p ? Wh3 : Wx3);
            #pragma unroll
            for (int hh = 0; hh < 2; ++hh) {
                const int idx = tid + hh * 256;   // 0..511
                const int r   = idx >> 3;         // 0..63
                const int c4  = idx & 7;
                float4 v = *(const float4*)(Wsrc + (long)(colN0 + r) * IN_DIM + k0 + c4 * 4);
                cvt_store4(&sW[buf][g][r][c4 * 4], v);
            }
        }
    };

    load_tile(0, 0);
    __syncthreads();

    for (int step = 0; step < 64; ++step) {
        const int buf = step & 1;
        if (step + 1 < 64) load_tile(buf ^ 1, step + 1);  // prefetch into other buffer

        const __bf16* aB = &sA[buf][0][0];
        const v16bf a0 = frag_a(aB, wm * 32,      lane);
        const v16bf a1 = frag_a(aB, wm * 32 + 16, lane);
        #pragma unroll
        for (int g = 0; g < 4; ++g) {
            const __bf16* bB = &sW[buf][g][0][0];
            const v16bf b0 = frag_b(bB, wn * 32,      lane);
            const v16bf b1 = frag_b(bB, wn * 32 + 16, lane);
            acc[g][0] = __builtin_amdgcn_wmma_f32_16x16x32_bf16(false, a0, false, b0, (short)0, acc[g][0], false, false);
            acc[g][1] = __builtin_amdgcn_wmma_f32_16x16x32_bf16(false, a0, false, b1, (short)0, acc[g][1], false, false);
            acc[g][2] = __builtin_amdgcn_wmma_f32_16x16x32_bf16(false, a1, false, b0, (short)0, acc[g][2], false, false);
            acc[g][3] = __builtin_amdgcn_wmma_f32_16x16x32_bf16(false, a1, false, b1, (short)0, acc[g][3], false, false);
        }
        __syncthreads();
    }

    // Fused LSTM epilogue. D layout: vgpr r, lane l -> M = r + 8*(l>=16), N = l&15.
    const long BH = (long)B_DIM * H_DIM;
    #pragma unroll
    for (int q = 0; q < 4; ++q) {
        const int  sm  = q >> 1, sn = q & 1;
        const int  col = colN0 + wn * 32 + sn * 16 + (lane & 15);
        const long rowBase = rowM0 + wm * 32 + sm * 16 + ((lane < 16) ? 0 : 8);
        const float vbi = bi[col], vbf = bf_[col], vbg = bg[col], vbo = bo[col];
        #pragma unroll
        for (int r = 0; r < 8; ++r) {
            const long row = rowBase + r;
            const float it = sigmoid_fast(acc[0][q][r] + vbi);
            const float ft = sigmoid_fast(acc[1][q][r] + vbf);
            const float gt = tanh_fast   (acc[2][q][r] + vbg);
            const float ot = sigmoid_fast(acc[3][q][r] + vbo);
            const float ct = ft * cPrev[row * H_DIM + col] + it * gt;
            const float ht = tanh_fast(ct) * ot;
            out[          row * H_DIM + col] = ht;   // h_t
            out[BH      + row * H_DIM + col] = ht;   // h_t (tuple repeat)
            out[2 * BH  + row * H_DIM + col] = ct;   // c_t
        }
    }
}

extern "C" void kernel_launch(void* const* d_in, const int* in_sizes, int n_in,
                              void* d_out, int out_size, void* d_ws, size_t ws_size,
                              hipStream_t stream) {
    const float* x     = (const float*)d_in[0];
    const float* hPrev = (const float*)d_in[1];
    const float* cPrev = (const float*)d_in[2];
    const float* W_ii  = (const float*)d_in[3];
    const float* W_hi  = (const float*)d_in[4];
    const float* b_i   = (const float*)d_in[5];
    const float* W_if  = (const float*)d_in[6];
    const float* W_hf  = (const float*)d_in[7];
    const float* b_f   = (const float*)d_in[8];
    const float* W_ig  = (const float*)d_in[9];
    const float* W_hg  = (const float*)d_in[10];
    const float* b_g   = (const float*)d_in[11];
    const float* W_io  = (const float*)d_in[12];
    const float* W_ho  = (const float*)d_in[13];
    const float* b_o   = (const float*)d_in[14];
    float* out = (float*)d_out;

    dim3 grid(B_DIM / 128, H_DIM / 64);   // 64 x 16 workgroups
    dim3 block(256);                      // 8 wave32
    lstm_fused_wmma<<<grid, block, 0, stream>>>(
        x, hPrev, cPrev,
        W_ii, W_if, W_ig, W_io,
        W_hi, W_hf, W_hg, W_ho,
        b_i, b_f, b_g, b_o,
        out);
}
